// MultiHeadAttention_53437983097091
// MI455X (gfx1250) — matrix operand order repaired and checked by reading, not compiled, not run on hardware
//
#include <hip/hip_runtime.h>

#define BATCH 4
#define SEQ   2048
#define DIM   1024
#define HEADS 16
#define HDIM  64

typedef __attribute__((ext_vector_type(16))) __bf16          v16bf;
typedef __attribute__((ext_vector_type(16))) unsigned short  v16us;
typedef __attribute__((ext_vector_type(8)))  unsigned short  us8;
typedef __attribute__((ext_vector_type(8)))  float           v8f;
typedef __attribute__((ext_vector_type(8)))  __bf16          v8bf;

#if __has_builtin(__builtin_amdgcn_tensor_load_to_lds) && __has_builtin(__builtin_amdgcn_s_wait_tensorcnt)
#define HAVE_TDM 1
typedef __attribute__((ext_vector_type(4))) unsigned int u32x4;
typedef __attribute__((ext_vector_type(8))) int          i32x8;
typedef __attribute__((ext_vector_type(4))) int          i32x4;
#else
#define HAVE_TDM 0
#endif

__device__ __forceinline__ unsigned short f2bf(float f) {
  return __builtin_bit_cast(unsigned short, (__bf16)f);   // single v_cvt
}

__device__ __forceinline__ v16bf mk16(us8 lo, us8 hi) {
  v16us t;
#pragma unroll
  for (int e = 0; e < 8; ++e) { t[e] = lo[e]; t[8 + e] = hi[e]; }
  return __builtin_bit_cast(v16bf, t);
}

__device__ __forceinline__ v8f zero8() {
  v8f z;
#pragma unroll
  for (int i = 0; i < 8; ++i) z[i] = 0.0f;
  return z;
}

#if HAVE_TDM
// TDM D# (ISA 8.3/8.4): 2D tile of a [1024 x 1024] bf16 tensor,
// tile_dim0=32 (contiguous cols), tile_dim1=64 (rows), row stride 1024.
__device__ __forceinline__ void tdm_load_w_tile(const unsigned short* gsrc,
                                                unsigned lds_off) {
  const unsigned long long ga = (unsigned long long)(size_t)gsrc;
  u32x4 g0;
  g0[0] = 1u;                                            // count=1, user D#
  g0[1] = lds_off;                                       // lds_addr (bytes)
  g0[2] = (unsigned)(ga & 0xFFFFFFFFull);                // global_addr[31:0]
  g0[3] = (unsigned)((ga >> 32) & 0x1FFFFFFull)          // global_addr[56:32]
        | (2u << 30);                                    // type=2 ("image")
  i32x8 g1;
  g1[0] = (int)(1u << 16);                               // data_size=1 (2B), mask=0
  g1[1] = (int)((unsigned)(DIM & 0xFFFF) << 16);         // tensor_dim0[15:0]
  g1[2] = (int)((unsigned)(DIM >> 16)                    // tensor_dim0[31:16]
        | ((unsigned)(DIM & 0xFFFF) << 16));             // tensor_dim1[15:0]
  g1[3] = (int)((unsigned)(DIM >> 16) | (32u << 16));    // tensor_dim1 hi | tile_dim0=32
  g1[4] = (int)64u;                                      // tile_dim1=64, tile_dim2=0
  g1[5] = (int)DIM;                                      // tensor_dim0_stride[31:0]
  g1[6] = 0;                                             // stride hi | dim1_stride lo
  g1[7] = 0;
  i32x4 z4; z4[0] = z4[1] = z4[2] = z4[3] = 0;
  i32x8 z8;
#pragma unroll
  for (int i = 0; i < 8; ++i) z8[i] = 0;
  __builtin_amdgcn_tensor_load_to_lds(g0, g1, z4, z4, z8, 0);
}
#endif

// ---------------------------------------------------------------------------
// Bulk f32 -> bf16 conversion (vectorized -> v_cvt_pk_bf16_f32)
// ---------------------------------------------------------------------------
__global__ __launch_bounds__(256) void cvt_bf16_kernel(const float* __restrict__ in,
                                                       unsigned short* __restrict__ out,
                                                       int n8)
{
  const int i = blockIdx.x * 256 + threadIdx.x;
  if (i >= n8) return;
  v8f f = *(const v8f*)(in + (size_t)i * 8);
  v8bf b = __builtin_convertvector(f, v8bf);
  *(us8*)(out + (size_t)i * 8) = __builtin_bit_cast(us8, b);
}

// ---------------------------------------------------------------------------
// Y = x @ W^T + b (bf16 in, bf16 out).
// transposed==0 : out layout [B,H,S,Hd]  (Q, K)
// transposed==1 : out layout [B,H,Hd,S]  (V)
// Block = 8 waves sharing one 64-feature N-group across 8 M-tiles, so the
// 64x32 W tile per k-step is staged ONCE into LDS by the Tensor Data Mover
// (double-buffered, s_wait_tensorcnt) and consumed by all 8 waves.
// ---------------------------------------------------------------------------
__global__ __launch_bounds__(256) void qkv_proj_kernel(
    const unsigned short* __restrict__ xb, const unsigned short* __restrict__ wb,
    const float* __restrict__ bias, unsigned short* __restrict__ out,
    int transposed)
{
#if HAVE_TDM
  __shared__ unsigned short sW[2][64 * 32];   // double-buffered W tile (8KB)
#endif
  const int tid  = threadIdx.x;
  const int wave = tid >> 5;
  const int lane = tid & 31;
  const int m0   = ((blockIdx.x >> 4) * 8 + wave) * 16;  // 8 M-tiles per block
  const int n0   = (blockIdx.x & 15) * 64;               // shared N-group

  const int lm   = lane & 15;
  const int lk8  = (lane >> 4) * 8;
  const int lk16 = (lane >> 4) * 16;

  v8f acc[4];
#pragma unroll
  for (int j = 0; j < 4; ++j) acc[j] = zero8();

  const unsigned short* xrow  = xb + (size_t)(m0 + lm) * DIM;
  const unsigned short* wtile = wb + (size_t)n0 * DIM;   // rows n0..n0+63

#if HAVE_TDM
  if (wave == 0)
    tdm_load_w_tile(wtile, (unsigned)(size_t)&sW[0][0]);
#else
  const unsigned short* wbase = wb + (size_t)(n0 + lm) * DIM + lk16;
#endif

  for (int kb = 0; kb < DIM; kb += 32) {
    __builtin_prefetch(xrow + kb + 32, 0, 3);
#if HAVE_TDM
    const int cur = (kb >> 5) & 1;
    if (wave == 0) {
      if (kb + 32 < DIM) {
        tdm_load_w_tile(wtile + kb + 32, (unsigned)(size_t)&sW[cur ^ 1][0]);
        __builtin_amdgcn_s_wait_tensorcnt(1);   // current tile complete
      } else {
        __builtin_amdgcn_s_wait_tensorcnt(0);
      }
    }
    __syncthreads();
#endif
    v16bf a = mk16(*(const us8*)(xrow + kb + lk16),
                   *(const us8*)(xrow + kb + lk16 + 8));
    v16bf bfr[4];
#pragma unroll
    for (int j = 0; j < 4; ++j) {
#if HAVE_TDM
      const unsigned short* wrow = &sW[cur][(j * 16 + lm) * 32 + lk16];
#else
      const unsigned short* wrow = wbase + (size_t)(j * 16) * DIM + kb;
#endif
      bfr[j] = mk16(*(const us8*)(wrow), *(const us8*)(wrow + 8));
    }
#pragma unroll
    for (int j = 0; j < 4; ++j)
      acc[j] = __builtin_amdgcn_wmma_f32_16x16x32_bf16(
          false, a, false, bfr[j], (short)0, acc[j], false, false);
#if HAVE_TDM
    __syncthreads();   // all waves done with sW[cur] before it is refilled
#endif
  }

  // bias + layout-aware bf16 store. C/D layout: n = lane&15, m = (lane>>4)*8 + v
#pragma unroll
  for (int j = 0; j < 4; ++j) {
    const int n  = n0 + j * 16 + lm;
    const float bv = bias[n];
#pragma unroll
    for (int v = 0; v < 8; ++v) {
      const int m = m0 + (lane >> 4) * 8 + v;
      const float y = acc[j][v] + bv;
      const int b = m >> 11, s = m & (SEQ - 1);
      const int h = n >> 6,  hd = n & (HDIM - 1);
      size_t idx;
      if (!transposed) idx = (((size_t)(b * HEADS + h) * SEQ) + s) * HDIM + hd;
      else             idx = (((size_t)(b * HEADS + h) * HDIM) + hd) * SEQ + s;
      out[idx] = f2bf(y);
    }
  }
}

// ---------------------------------------------------------------------------
// One block per (batch, 16-query tile), 256 threads = 8 waves (2 per SIMD32,
// since 320KB LDS means one block per WGP). Loops all 16 heads.
// LDS: scores f32 16x2048 (128KB) + attn-mean f32 16x2048 (128KB)
//      + probs bf16 16x2048 (64KB)  == 320KB (CDNA5 per-workgroup max).
// ---------------------------------------------------------------------------
__global__ __launch_bounds__(256) void attn_kernel(
    const unsigned short* __restrict__ Q,   // [B,H,S,Hd] bf16
    const unsigned short* __restrict__ Kt,  // [B,H,S,Hd] bf16
    const unsigned short* __restrict__ Vt,  // [B,H,Hd,S] bf16
    float* __restrict__ ctx,                // [B,S,D] f32
    float* __restrict__ attnm)              // [B,S,S] f32
{
  extern __shared__ char smem[];
  float*          sS = (float*)smem;                                   // scores/exp + scratch
  float*          sA = (float*)(smem + (size_t)16 * SEQ * 4);          // attn-mean acc
  unsigned short* sP = (unsigned short*)(smem + (size_t)32 * SEQ * 4); // probs
  float*          sRed = (float*)sP;                                   // overlay scratch

  const int tid  = threadIdx.x;
  const int wave = tid >> 5;
  const int lane = tid & 31;
  const int bb   = blockIdx.x >> 7;          // batch
  const int q0   = (blockIdx.x & 127) << 4;  // query tile base

  const int lm   = lane & 15;
  const int lk8  = (lane >> 4) * 8;
  const int lk16 = (lane >> 4) * 16;

  for (int i = tid; i < 16 * SEQ; i += 256) sA[i] = 0.0f;
  __syncthreads();

  for (int h = 0; h < HEADS; ++h) {
    const unsigned short* qB = Q  + ((size_t)(bb * HEADS + h) * SEQ + q0) * HDIM;
    const unsigned short* kB = Kt +  (size_t)(bb * HEADS + h) * SEQ * HDIM;
    const unsigned short* vB = Vt +  (size_t)(bb * HEADS + h) * HDIM * SEQ;

    // --- GEMM1: scores = (Q_tile @ K^T) * rsqrt(Hd). 256 keys per wave.
    const unsigned short* qr = qB + (size_t)lm * HDIM;
    v16bf a0 = mk16(*(const us8*)(qr + lk16),      *(const us8*)(qr + lk16 + 8));
    v16bf a1 = mk16(*(const us8*)(qr + 32 + lk16), *(const us8*)(qr + 32 + lk16 + 8));

#pragma unroll 2
    for (int i = 0; i < 16; ++i) {
      const int kt = (wave << 8) + (i << 4);
      const unsigned short* kr = kB + (size_t)(kt + lm) * HDIM;
      v16bf b0 = mk16(*(const us8*)(kr + lk16),      *(const us8*)(kr + lk16 + 8));
      v16bf b1 = mk16(*(const us8*)(kr + 32 + lk16), *(const us8*)(kr + 32 + lk16 + 8));
      v8f c = zero8();
      c = __builtin_amdgcn_wmma_f32_16x16x32_bf16(false, a0, false, b0, (short)0, c, false, false);
      c = __builtin_amdgcn_wmma_f32_16x16x32_bf16(false, a1, false, b1, (short)0, c, false, false);
#pragma unroll
      for (int v = 0; v < 8; ++v)
        sS[(size_t)((lane >> 4) * 8 + v) * SEQ + kt + lm] = c[v] * 0.125f;
    }
    __syncthreads();

    // --- softmax: 16 threads/row, 128 cols each
    const int r = tid >> 4, pp = tid & 15;
    float* row = sS + (size_t)r * SEQ + pp * 128;
    float mx = -3.4e38f;
    for (int i = 0; i < 128; ++i) mx = fmaxf(mx, row[i]);
    sRed[tid] = mx;
    __syncthreads();
    float rmax = -3.4e38f;
#pragma unroll
    for (int j = 0; j < 16; ++j) rmax = fmaxf(rmax, sRed[r * 16 + j]);
    float sm = 0.0f;
    for (int i = 0; i < 128; ++i) {       // cache exp in place (same thread)
      const float e = __expf(row[i] - rmax);
      row[i] = e;
      sm += e;
    }
    sRed[256 + tid] = sm;
    __syncthreads();
    float rsum = 0.0f;
#pragma unroll
    for (int j = 0; j < 16; ++j) rsum += sRed[256 + r * 16 + j];
    const float rinv = 1.0f / rsum;
    __syncthreads();  // all sRed reads done before probs overwrite the overlay

    unsigned short* prow = sP + (size_t)r * SEQ + pp * 128;
    float*          arow = sA + (size_t)r * SEQ + pp * 128;
    for (int i = 0; i < 128; ++i) {
      const float pv = row[i] * rinv;
      prow[i] = f2bf(pv);
      arow[i] += pv * 0.0625f;            // 1/HEADS
    }
    __syncthreads();

    // --- GEMM2: ctx_tile = probs @ V.
    // wave = kh*4 + nt : nt picks 16 of 64 hd cols, kh picks half of the keys.
    const int nt = wave & 3;
    const int kh = wave >> 2;
    v8f acc = zero8();
    const unsigned short* vrow = vB + (size_t)(nt * 16 + lm) * SEQ;
    const int kbeg = kh * (SEQ / 2);
    for (int k0 = kbeg; k0 < kbeg + SEQ / 2; k0 += 64) {
      const unsigned short* pr = sP + (size_t)lm * SEQ + k0;
      v16bf pa0 = mk16(*(const us8*)(pr + lk16),      *(const us8*)(pr + lk16 + 8));
      v16bf pa1 = mk16(*(const us8*)(pr + 32 + lk16), *(const us8*)(pr + 32 + lk16 + 8));
      v16bf vb0 = mk16(*(const us8*)(vrow + k0 + lk16),      *(const us8*)(vrow + k0 + lk16 + 8));
      v16bf vb1 = mk16(*(const us8*)(vrow + k0 + 32 + lk16), *(const us8*)(vrow + k0 + 32 + lk16 + 8));
      acc = __builtin_amdgcn_wmma_f32_16x16x32_bf16(false, pa0, false, vb0, (short)0, acc, false, false);
      acc = __builtin_amdgcn_wmma_f32_16x16x32_bf16(false, pa1, false, vb1, (short)0, acc, false, false);
    }
    // combine the two K-halves through sS (free during GEMM2)
    if (kh == 1) {
#pragma unroll
      for (int v = 0; v < 8; ++v)
        sS[nt * 256 + ((lane >> 4) * 8 + v) * 16 + lm] = acc[v];
    }
    __syncthreads();
    if (kh == 0) {
#pragma unroll
      for (int v = 0; v < 8; ++v) {
        const int m = q0 + (lane >> 4) * 8 + v;
        const int n = h * HDIM + nt * 16 + lm;
        ctx[((size_t)bb * SEQ + m) * DIM + n] =
            acc[v] + sS[nt * 256 + ((lane >> 4) * 8 + v) * 16 + lm];
      }
    }
    __syncthreads();
  }

  // single non-atomic write of the head-averaged attention tile
  for (int i = tid; i < 16 * SEQ; i += 256) {
    const int r = i >> 11, c = i & (SEQ - 1);
    attnm[((size_t)bb * SEQ + q0 + r) * SEQ + c] = sA[i];
  }
}

// pooled[b,d] = mean_s ctx[b,s,d]
__global__ __launch_bounds__(256) void pool_kernel(const float* __restrict__ ctx,
                                                   float* __restrict__ pooled)
{
  const int b = blockIdx.x >> 2;
  const int d = ((blockIdx.x & 3) << 8) + threadIdx.x;
  const float* p = ctx + (size_t)b * SEQ * DIM + d;
  float s = 0.0f;
  for (int i = 0; i < SEQ; ++i) s += p[(size_t)i * DIM];
  pooled[b * DIM + d] = s * (1.0f / (float)SEQ);
}

// out[b,n] = bo[n] + pooled[b,:] . wo[n,:]
__global__ __launch_bounds__(256) void out_kernel(const float* __restrict__ pooled,
                                                  const float* __restrict__ wo,
                                                  const float* __restrict__ bo,
                                                  float* __restrict__ out)
{
  const int b = blockIdx.x >> 2;
  const int n = ((blockIdx.x & 3) << 8) + threadIdx.x;
  const float* pr = pooled + b * DIM;
  const float* wr = wo + (size_t)n * DIM;
  float s = bo[n];
  for (int d = 0; d < DIM; ++d) s = fmaf(pr[d], wr[d], s);
  out[b * DIM + n] = s;
}

extern "C" void kernel_launch(void* const* d_in, const int* in_sizes, int n_in,
                              void* d_out, int out_size, void* d_ws, size_t ws_size,
                              hipStream_t stream)
{
  const float* x  = (const float*)d_in[0];
  const float* wq = (const float*)d_in[1];
  const float* bq = (const float*)d_in[2];
  const float* wk = (const float*)d_in[3];
  const float* bk = (const float*)d_in[4];
  const float* wv = (const float*)d_in[5];
  const float* bv = (const float*)d_in[6];
  const float* wo = (const float*)d_in[7];
  const float* bo = (const float*)d_in[8];
  float* out = (float*)d_out;

  const size_t MTOT = (size_t)BATCH * SEQ;            // 8192
  unsigned short* xb  = (unsigned short*)d_ws;        // 16 MB bf16 copy of x
  unsigned short* wqb = xb  + MTOT * DIM;             // 2 MB each
  unsigned short* wkb = wqb + (size_t)DIM * DIM;
  unsigned short* wvb = wkb + (size_t)DIM * DIM;
  unsigned short* Qb  = wvb + (size_t)DIM * DIM;      // 16 MB each
  unsigned short* Kb  = Qb  + MTOT * DIM;
  unsigned short* Vtb = Kb  + MTOT * DIM;
  float* ctx    = (float*)(Vtb + MTOT * DIM);         // 32 MB f32
  float* pooled = ctx + MTOT * DIM;

  // one-time f32 -> bf16 conversions (bandwidth-trivial)
  const int nx8 = (int)(MTOT * DIM / 8);              // 1,048,576
  const int nw8 = DIM * DIM / 8;                      // 131,072
  cvt_bf16_kernel<<<dim3(nx8 / 256), dim3(256), 0, stream>>>(x,  xb,  nx8);
  cvt_bf16_kernel<<<dim3(nw8 / 256), dim3(256), 0, stream>>>(wq, wqb, nw8);
  cvt_bf16_kernel<<<dim3(nw8 / 256), dim3(256), 0, stream>>>(wk, wkb, nw8);
  cvt_bf16_kernel<<<dim3(nw8 / 256), dim3(256), 0, stream>>>(wv, wvb, nw8);

  // QKV projections (bf16 WMMA, f32 accumulate, TDM-staged W tiles)
  qkv_proj_kernel<<<dim3(1024), dim3(256), 0, stream>>>(xb, wqb, bq, Qb,  0);
  qkv_proj_kernel<<<dim3(1024), dim3(256), 0, stream>>>(xb, wkb, bk, Kb,  0);
  qkv_proj_kernel<<<dim3(1024), dim3(256), 0, stream>>>(xb, wvb, bv, Vtb, 1);

  // Attention: 320 KB LDS per workgroup (CDNA5 max)
  const size_t shmem = (size_t)32 * SEQ * 4 + (size_t)16 * SEQ * 2;  // 327680
  (void)hipFuncSetAttribute((const void*)attn_kernel,
                            hipFuncAttributeMaxDynamicSharedMemorySize, (int)shmem);
  attn_kernel<<<dim3(BATCH * (SEQ / 16)), dim3(256), shmem, stream>>>(
      Qb, Kb, Vtb, ctx, out + BATCH * DIM);

  pool_kernel<<<dim3(16), dim3(256), 0, stream>>>(ctx, pooled);
  out_kernel<<<dim3(16), dim3(256), 0, stream>>>(pooled, wo, bo, out);
}
